// NeuroSAT_27144193311174
// MI455X (gfx1250) — matrix operand order repaired and checked by reading, not compiled, not run on hardware
//
#include <hip/hip_runtime.h>

typedef __attribute__((ext_vector_type(16))) __bf16 v16bf;
typedef __attribute__((ext_vector_type(8)))  float  v8f;
typedef __attribute__((ext_vector_type(2)))  __bf16 bf16x2;

#define HDIM 128

// ---------------------------------------------------------------------------
// WMMA GEMM: out = act(A[M,K]_bf16 @ W[K,Ntot] + bias), W pre-transposed as
// Wt[Ntot,K] bf16. Block = 256 threads (8 waves), tile = 256M x 128N.
// Each wave owns a 32x128 stripe: 2 A fragments x 8 B fragments -> 16 WMMAs
// per K-step, so every LDS B-fragment load feeds two v_wmma ops.
// outB != null -> relu + bf16 store; else f32 store with optional accumulate.
// M must be a multiple of 16 (second 16-row half is guarded per wave).
// ---------------------------------------------------------------------------
__global__ __launch_bounds__(256)
void wmma_gemm_bf16(const __bf16* __restrict__ A,
                    const __bf16* __restrict__ Wt,
                    const float*  __restrict__ bias,
                    float* __restrict__ outF,
                    __bf16* __restrict__ outB,
                    int M, int K, int Ntot, int accumulate)
{
    const int mTile = blockIdx.x * 256;
    const int nTile = blockIdx.y * 128;
    const int wave  = threadIdx.x >> 5;   // wave32
    const int lane  = threadIdx.x & 31;

    __shared__ __bf16 sW[128 * 256];      // Wt tile: [128 n][K k], K <= 256

    // cooperative stage of Wt rows [nTile, nTile+128)
    {
        const int totDw = (128 * K) >> 1;
        const uint32_t* src = (const uint32_t*)(Wt + (size_t)nTile * K);
        uint32_t* dst = (uint32_t*)sW;
        for (int i = threadIdx.x; i < totDw; i += 256) dst[i] = src[i];
    }
    __syncthreads();

    const int m0 = mTile + wave * 32;     // first 16-row half
    const int m1 = m0 + 16;               // second 16-row half
    if (m0 >= M) return;                  // wave-uniform guard
    const bool half2 = (m1 < M);          // wave-uniform
    const int lhi = lane >> 4;            // 0 or 1
    const int lm  = lane & 15;

    v8f acc[16];
    #pragma unroll
    for (int t = 0; t < 16; ++t)
        #pragma unroll
        for (int v = 0; v < 8; ++v) acc[t][v] = 0.0f;

    for (int k0 = 0; k0 < K; k0 += 32) {
        // A fragments: lane<16 rows M=lm hold K {0..7,16..23}; lane>=16 hold
        // K {8..15,24..31}; each dword pair is contiguous -> b128 loads.
        v16bf a0, a1;
        {
            const __bf16* arow = A + (size_t)(m0 + lm) * K + k0;
            #pragma unroll
            for (int v = 0; v < 8; ++v) {
                const int kk = (v < 4) ? (2 * v + 8 * lhi)
                                       : (16 + 2 * (v - 4) + 8 * lhi);
                uint32_t w = *(const uint32_t*)(arow + kk);
                bf16x2 p = __builtin_bit_cast(bf16x2, w);
                a0[2 * v]     = p[0];
                a0[2 * v + 1] = p[1];
            }
        }
        if (half2) {
            const __bf16* arow = A + (size_t)(m1 + lm) * K + k0;
            #pragma unroll
            for (int v = 0; v < 8; ++v) {
                const int kk = (v < 4) ? (2 * v + 8 * lhi)
                                       : (16 + 2 * (v - 4) + 8 * lhi);
                uint32_t w = *(const uint32_t*)(arow + kk);
                bf16x2 p = __builtin_bit_cast(bf16x2, w);
                a1[2 * v]     = p[0];
                a1[2 * v + 1] = p[1];
            }
        } else {
            #pragma unroll
            for (int v = 0; v < 16; ++v) a1[v] = (__bf16)0.0f;
        }

        #pragma unroll
        for (int t = 0; t < 8; ++t) {
            // B fragment: column n = 16t+lm, lane<16 K {0..7,16..23}, lane>=16 K {8..15,24..31}
            v16bf b;
            const __bf16* brow = sW + (size_t)(16 * t + lm) * K + k0;
            #pragma unroll
            for (int v = 0; v < 8; ++v) {
                const int kk = (v < 4) ? (2 * v + 8 * lhi) : (16 + 2 * (v - 4) + 8 * lhi);
                uint32_t w = *(const uint32_t*)(brow + kk);
                bf16x2 p = __builtin_bit_cast(bf16x2, w);
                b[2 * v]     = p[0];
                b[2 * v + 1] = p[1];
            }
            acc[t] = __builtin_amdgcn_wmma_f32_16x16x32_bf16(
                false, a0, false, b, (short)0, acc[t], false, false);
            acc[8 + t] = __builtin_amdgcn_wmma_f32_16x16x32_bf16(
                false, a1, false, b, (short)0, acc[8 + t], false, false);
        }
    }

    // Epilogue: D layout lane<16 -> (M=v, N=lm), lane>=16 -> (M=v+8, N=lm)
    #pragma unroll
    for (int h = 0; h < 2; ++h) {
        if (h == 1 && !half2) break;
        const int mBase = (h == 0) ? m0 : m1;
        #pragma unroll
        for (int t = 0; t < 8; ++t) {
            const int n  = nTile + 16 * t + lm;
            const float bv = bias ? bias[n] : 0.0f;
            #pragma unroll
            for (int v = 0; v < 8; ++v) {
                const int m = mBase + v + 8 * lhi;
                float val = acc[8 * h + t][v] + bv;
                const size_t idx = (size_t)m * Ntot + n;
                if (outB) {
                    val = val > 0.0f ? val : 0.0f;
                    outB[idx] = (__bf16)val;
                } else {
                    if (accumulate) val += outF[idx];
                    outF[idx] = val;
                }
            }
        }
    }
}

// out[sidx[e]+sOff] += msg[gidx[e]+gOff], per H=128 element, fp32 atomics
__global__ void gather_scatter_add(const float* __restrict__ msg,
                                   const int* __restrict__ gidx,
                                   const int* __restrict__ sidx,
                                   float* __restrict__ out,
                                   int E, int gOff, int sOff)
{
    long long i = (long long)blockIdx.x * blockDim.x + threadIdx.x;
    if (i >= (long long)E * HDIM) return;
    const int e = (int)(i >> 7);
    const int h = (int)(i & 127);
    const float v = msg[(size_t)(gidx[e] + gOff) * HDIM + h];
    __hip_atomic_fetch_add(&out[(size_t)(sidx[e] + sOff) * HDIM + h], v,
                           __ATOMIC_RELAXED, __HIP_MEMORY_SCOPE_AGENT);
}

// LSTM pointwise: g[N,4H] (i,f,g,o), cell state cst[N,H] updated in place,
// h[N,H] written (lives in d_out).
__global__ void lstm_pointwise(const float* __restrict__ g,
                               float* __restrict__ cst,
                               float* __restrict__ h, long long N)
{
    long long i = (long long)blockIdx.x * blockDim.x + threadIdx.x;
    if (i >= N * HDIM) return;
    const long long row = i >> 7;
    const int hh = (int)(i & 127);
    const float* gr = g + row * (4 * HDIM);
    const float gi = gr[hh];
    const float gf = gr[HDIM + hh];
    const float gg = gr[2 * HDIM + hh];
    const float go = gr[3 * HDIM + hh];
    const float si = 1.0f / (1.0f + __expf(-gi));
    const float sf = 1.0f / (1.0f + __expf(-gf));
    const float so = 1.0f / (1.0f + __expf(-go));
    const float c  = sf * cst[i] + si * tanhf(gg);
    cst[i] = c;
    h[i] = so * tanhf(c);
}

__global__ void f32_to_bf16(const float* __restrict__ in,
                            __bf16* __restrict__ out, long long n)
{
    long long i = (long long)blockIdx.x * blockDim.x + threadIdx.x;
    if (i < n) out[i] = (__bf16)in[i];
}

// W[Ktot,N] rows [kOff,kOff+K) -> out[N,K] bf16 (transpose + convert)
__global__ void weight_tc(const float* __restrict__ W,
                          __bf16* __restrict__ out, int K, int N, int kOff)
{
    long long i = (long long)blockIdx.x * blockDim.x + threadIdx.x;
    if (i >= (long long)K * N) return;
    const int k = (int)(i / N);
    const int n = (int)(i % N);
    out[(size_t)n * K + k] = (__bf16)W[(size_t)(k + kOff) * N + n];
}

extern "C" void kernel_launch(void* const* d_in, const int* in_sizes, int n_in,
                              void* d_out, int out_size, void* d_ws, size_t ws_size,
                              hipStream_t stream)
{
    (void)in_sizes; (void)n_in; (void)out_size; (void)ws_size;
    const int V = 50000, L = 100000, C = 210000, E = 315000, H = HDIM, R = 4;

    const float* l_emb = (const float*)d_in[0];
    const float* c_emb = (const float*)d_in[1];
    const int* eps  = (const int*)d_in[2];
    const int* epd  = (const int*)d_in[3];
    const int* ens  = (const int*)d_in[4];
    const int* endd = (const int*)d_in[5];
    const float* lmlp_w[3] = {(const float*)d_in[6], (const float*)d_in[8],  (const float*)d_in[10]};
    const float* lmlp_b[3] = {(const float*)d_in[7], (const float*)d_in[9],  (const float*)d_in[11]};
    const float* cmlp_w[3] = {(const float*)d_in[12], (const float*)d_in[14], (const float*)d_in[16]};
    const float* cmlp_b[3] = {(const float*)d_in[13], (const float*)d_in[15], (const float*)d_in[17]};
    const float* l_w_ih = (const float*)d_in[18];
    const float* l_w_hh = (const float*)d_in[19];
    const float* l_b    = (const float*)d_in[20];
    const float* c_w_ih = (const float*)d_in[21];
    const float* c_w_hh = (const float*)d_in[22];
    const float* c_b    = (const float*)d_in[23];

    float* out_lh = (float*)d_out;                 // [L,H]
    float* out_ch = out_lh + (size_t)L * H;        // [C,H]

    // ---- bump allocator over d_ws ----
    char* p = (char*)d_ws;
    auto alloc = [&](size_t bytes) -> void* {
        void* r = (void*)p;
        p += (bytes + 255) & ~(size_t)255;
        return r;
    };
    __bf16* lh_bf   = (__bf16*)alloc((size_t)L * H * 2);
    __bf16* ch_bf   = (__bf16*)alloc((size_t)C * H * 2);
    __bf16* tA      = (__bf16*)alloc((size_t)C * H * 2);
    __bf16* tB      = (__bf16*)alloc((size_t)C * H * 2);
    __bf16* l2c_bf  = (__bf16*)alloc((size_t)C * H * 2);
    __bf16* c2l_bf  = (__bf16*)alloc((size_t)L * H * 2);
    float*  msg_f   = (float*)alloc((size_t)C * H * 4);
    float*  l2c_f   = (float*)alloc((size_t)C * H * 4);
    float*  c2l_f   = (float*)alloc((size_t)L * H * 4);
    float*  gbuf    = (float*)alloc((size_t)C * 4 * H * 4);
    float*  l_cst   = (float*)alloc((size_t)L * H * 4);
    float*  c_cst   = (float*)alloc((size_t)C * H * 4);
    __bf16* lw_t[3]; __bf16* cw_t[3];
    for (int j = 0; j < 3; ++j) lw_t[j] = (__bf16*)alloc((size_t)H * H * 2);
    for (int j = 0; j < 3; ++j) cw_t[j] = (__bf16*)alloc((size_t)H * H * 2);
    __bf16* l_ih_a_t = (__bf16*)alloc((size_t)512 * H * 2);
    __bf16* l_ih_b_t = (__bf16*)alloc((size_t)512 * H * 2);
    __bf16* l_hh_t   = (__bf16*)alloc((size_t)512 * H * 2);
    __bf16* c_ih_t   = (__bf16*)alloc((size_t)512 * H * 2);
    __bf16* c_hh_t   = (__bf16*)alloc((size_t)512 * H * 2);

    // ---- launch helpers ----
    auto gemm = [&](const __bf16* A, const __bf16* Wt, const float* bias,
                    float* oF, __bf16* oB, int M, int K, int Ntot, int acc) {
        dim3 grid((M + 255) / 256, Ntot / 128);
        wmma_gemm_bf16<<<grid, 256, 0, stream>>>(A, Wt, bias, oF, oB, M, K, Ntot, acc);
    };
    auto cvt = [&](const float* in, __bf16* out, long long n) {
        f32_to_bf16<<<(int)((n + 255) / 256), 256, 0, stream>>>(in, out, n);
    };
    auto wtc = [&](const float* W, __bf16* out, int K, int N, int kOff) {
        long long n = (long long)K * N;
        weight_tc<<<(int)((n + 255) / 256), 256, 0, stream>>>(W, out, K, N, kOff);
    };
    auto scat = [&](const float* msg, const int* g, const int* s, float* out,
                    int gOff, int sOff) {
        long long n = (long long)E * H;
        gather_scatter_add<<<(int)((n + 255) / 256), 256, 0, stream>>>(
            msg, g, s, out, E, gOff, sOff);
    };
    auto lstm = [&](const float* g, float* cst, float* h, long long N) {
        long long n = N * H;
        lstm_pointwise<<<(int)((n + 255) / 256), 256, 0, stream>>>(g, cst, h, N);
    };

    // ---- one-time per launch: weights -> bf16 transposed, init states ----
    for (int j = 0; j < 3; ++j) wtc(lmlp_w[j], lw_t[j], H, H, 0);
    for (int j = 0; j < 3; ++j) wtc(cmlp_w[j], cw_t[j], H, H, 0);
    wtc(l_w_ih, l_ih_a_t, H, 512, 0);     // w_ih[:H]   (acts on c2l)
    wtc(l_w_ih, l_ih_b_t, H, 512, H);     // w_ih[H:2H] (acts on flip)
    wtc(l_w_hh, l_hh_t, H, 512, 0);
    wtc(c_w_ih, c_ih_t, H, 512, 0);
    wtc(c_w_hh, c_hh_t, H, 512, 0);

    hipMemcpyAsync(out_lh, l_emb, (size_t)L * H * 4, hipMemcpyDeviceToDevice, stream);
    hipMemcpyAsync(out_ch, c_emb, (size_t)C * H * 4, hipMemcpyDeviceToDevice, stream);
    hipMemsetAsync(l_cst, 0, (size_t)L * H * 4, stream);
    hipMemsetAsync(c_cst, 0, (size_t)C * H * 4, stream);

    for (int r = 0; r < R; ++r) {
        // snapshot old hidden states in bf16 (used by MLPs, flip, and w_hh GEMMs)
        cvt(out_lh, lh_bf, (long long)L * H);
        cvt(out_ch, ch_bf, (long long)C * H);

        // literal MLP3 -> l_msg (fp32)
        gemm(lh_bf, lw_t[0], lmlp_b[0], nullptr, tA, L, H, H, 0);
        gemm(tA,    lw_t[1], lmlp_b[1], nullptr, tB, L, H, H, 0);
        gemm(tB,    lw_t[2], lmlp_b[2], msg_f, nullptr, L, H, H, 0);
        // l2c = seg_sum(pos_msg[src]->dst) + seg_sum(neg_msg[src]->dst)
        hipMemsetAsync(l2c_f, 0, (size_t)C * H * 4, stream);
        scat(msg_f, eps, epd, l2c_f, 0, 0);
        scat(msg_f, ens, endd, l2c_f, V, 0);

        // clause MLP3 -> c_msg (fp32)
        gemm(ch_bf, cw_t[0], cmlp_b[0], nullptr, tA, C, H, H, 0);
        gemm(tA,    cw_t[1], cmlp_b[1], nullptr, tB, C, H, H, 0);
        gemm(tB,    cw_t[2], cmlp_b[2], msg_f, nullptr, C, H, H, 0);
        // c2l: pos rows [0,V), neg rows [V,2V)
        hipMemsetAsync(c2l_f, 0, (size_t)L * H * 4, stream);
        scat(msg_f, epd, eps, c2l_f, 0, 0);
        scat(msg_f, endd, ens, c2l_f, 0, V);

        cvt(l2c_f, l2c_bf, (long long)C * H);
        cvt(c2l_f, c2l_bf, (long long)L * H);

        // clause LSTM: g = l2c @ c_w_ih + c_h @ c_w_hh + c_b
        gemm(l2c_bf, c_ih_t, c_b, gbuf, nullptr, C, H, 512, 0);
        gemm(ch_bf,  c_hh_t, nullptr, gbuf, nullptr, C, H, 512, 1);
        lstm(gbuf, c_cst, out_ch, C);

        // literal LSTM: g = c2l @ w_ih[:H] + flip @ w_ih[H:] + l_h @ w_hh + l_b
        gemm(c2l_bf, l_ih_a_t, l_b, gbuf, nullptr, L, H, 512, 0);
        // flip rows [0,V) = old l_h rows [V,2V); flip rows [V,2V) = old l_h rows [0,V)
        gemm(lh_bf + (size_t)V * H, l_ih_b_t, nullptr, gbuf, nullptr, V, H, 512, 1);
        gemm(lh_bf, l_ih_b_t, nullptr, gbuf + (size_t)V * 512, nullptr, V, H, 512, 1);
        gemm(lh_bf, l_hh_t, nullptr, gbuf, nullptr, L, H, 512, 1);
        lstm(gbuf, l_cst, out_lh, L);
    }
}
